// MAF_74423193305609
// MI455X (gfx1250) — matrix operand order repaired and checked by reading, not compiled, not run on hardware
//
#include <hip/hip_runtime.h>
#include <hip/hip_bf16.h>

// ---------------------------------------------------------------------------
// MAF forward pass for MI455X / gfx1250.
// Bandwidth-bound: fp32 weights streamed (once, and ONLY the unmasked half --
// MADE masks are thermometer masks in k, so the k-loop is clamped per wave),
// converted to bf16 in flight, consumed by v_wmma_f32_16x16x32_bf16.
// A-fragments are software-pipelined (rotate-by-one) so no WMMA waits on a
// fresh ds_load; schedule pinned with sched_group_barrier.
// ---------------------------------------------------------------------------

typedef __attribute__((ext_vector_type(16))) __bf16 v16bf;
typedef __attribute__((ext_vector_type(8)))  float  v8f;
typedef __attribute__((ext_vector_type(4)))  int    v4i;

#define BATCH   100
#define MPAD    112          // 7 tiles of 16
#define MTILES  7
#define DIMZ    1024
#define HID     2048
#define KSPLIT  8
#define LDS_PAD 8            // bf16 row padding (pitch 528B -> bank shift of 4 dwords)

// MASK semantics (k = weight row, n = weight col), all equivalent to k < T(n):
//   MASK==0 : (n>>1) >  k          ->  T = n>>1
//   MASK==1 : (n>>1) >= (k>>1)     ->  T = (n & ~1) + 2
//   MASK==3 :  n     >= (k>>1)     ->  T = 2n + 2
template <int MASK>
__device__ __forceinline__ int mask_thr(int n)
{
    if constexpr (MASK == 0) return n >> 1;
    else if constexpr (MASK == 1) return (n & ~1) + 2;
    else return 2 * n + 2;
}

template <int K, int N, int MASK>
__launch_bounds__(256)
__global__ void made_gemm_kernel(const __bf16* __restrict__ Ab,  // [MPAD, K] bf16, zero-padded
                                 const float*  __restrict__ W,   // [K, N]    fp32
                                 float*        __restrict__ Cp)  // [BATCH, N] fp32 partial (atomic)
{
    constexpr int Kc    = K / KSPLIT;
    constexpr int PITCH = Kc + LDS_PAD;
    constexpr int ROWB  = Kc / 8;        // b128 transfers per row

    const int kc0 = blockIdx.y * Kc;

    // ---- Block-level dead-chunk skip: everything above the block's max
    // threshold is masked to zero; those weights are never read at all.
    const int tblk = mask_thr<MASK>(blockIdx.x * 128 + 127);
    if (kc0 >= tblk) return;                     // uniform across the block
    const int kcols = min(Kc, ((tblk - kc0 + 31) & ~31));   // live cols, mult of 32

    extern __shared__ char smem_raw[];
    __bf16* sA = (__bf16*)smem_raw;

    const int tid  = threadIdx.x;
    const int lane = tid & 31;
    const int wave = tid >> 5;

    // ---- Stage A chunk [MPAD x kcols] bf16 into LDS via async LDS copies ----
    {
        const char*    Abase = (const char*)Ab + (size_t)kc0 * 2;
        const unsigned sbase = (unsigned)(uintptr_t)sA;
        for (int t = tid; t < MPAD * ROWB; t += 256) {   // exact multiple of 256
            const int m = t / ROWB;                      // ROWB is a power of two
            const int c = t - m * ROWB;
            if (c * 8 < kcols) {                         // per-lane EXEC predication
                const unsigned lds_off = sbase + (unsigned)(m * PITCH + c * 8) * 2u;
                const char*    src     = Abase + ((size_t)m * K + (size_t)c * 8) * 2;
                asm volatile("global_load_async_to_lds_b128 %0, %1, off"
                             :: "v"(lds_off), "v"(src) : "memory");
            }
        }
        asm volatile("s_wait_asynccnt 0x0" ::: "memory");
    }
    __syncthreads();

    // ---- Per-wave column strip ----
    const int n0   = blockIdx.x * 128 + wave * 16;
    const int cn   = n0 + (lane & 15);          // this lane's output column
    const int hiK  = (lane >> 4);               // 0: lanes 0-15, 1: lanes 16-31
    const int koff = hiK * 8;                   // B fragment: hi lanes hold K+8..K+15, K+24..K+31

    // Per-wave k clamp: rows >= T(strip max col) are fully masked for this wave.
    const int twav = mask_thr<MASK>(n0 + 15);
    const int kend = min(kcols, max(0, ((twav - kc0 + 31) & ~31)));

    if (kend <= 0) return;                       // whole strip masked out

    union AU { v4i q[2]; v16bf v; };
    const __bf16* aw = sA + (lane & 15) * PITCH + hiK * 8;   // wave's A base

    // Prologue: preload all 7 fragments for kk = 0.
    AU af[MTILES];
    #pragma unroll
    for (int mt = 0; mt < MTILES; ++mt) {
        const __bf16* arow = aw + mt * 16 * PITCH;
        af[mt].q[0] = *(const v4i*)(arow);
        af[mt].q[1] = *(const v4i*)(arow + 16);
    }

    v8f acc[MTILES] = {};

    for (int kk = 0; kk < kend; kk += 32) {
        const int krow0 = kc0 + kk + koff;

        // B fragment: W[krow0 .. krow0+7, krow0+16 .. krow0+23, cn], masked, fp32 -> bf16.
        v16bf bfrag;
        #pragma unroll
        for (int t = 0; t < 16; ++t) {
            const int krow = krow0 + t + (t & 8);
            float w = W[(size_t)krow * N + cn];
            const bool keep = krow < mask_thr<MASK>(cn);
            bfrag[t] = (__bf16)(keep ? w : 0.0f);
        }

        // Rotate-by-one pipeline: WMMA consumes the fragment loaded in the
        // previous iteration; immediately refill it for the next one. Last
        // iteration harmlessly reloads the current k (clamped, in-bounds).
        const int kkn = (kk + 32 < kend) ? (kk + 32) : kk;
        #pragma unroll
        for (int mt = 0; mt < MTILES; ++mt) {
            acc[mt] = __builtin_amdgcn_wmma_f32_16x16x32_bf16(
                          false, af[mt].v, false, bfrag, (short)0, acc[mt], false, false);
            const __bf16* arow = aw + mt * 16 * PITCH + kkn;
            af[mt].q[0] = *(const v4i*)(arow);
            af[mt].q[1] = *(const v4i*)(arow + 16);
        }

        // Pin the steady-state schedule: weights first (latency hidden under
        // the WMMA chain), then 7 x { 1 WMMA, 2 DS reads }. VALU floats free.
        __builtin_amdgcn_sched_group_barrier(0x020, 16, 0);  // VMEM reads
        __builtin_amdgcn_sched_group_barrier(0x008, 1, 0);   // WMMA
        __builtin_amdgcn_sched_group_barrier(0x100, 2, 0);   // DS read x2
        __builtin_amdgcn_sched_group_barrier(0x008, 1, 0);
        __builtin_amdgcn_sched_group_barrier(0x100, 2, 0);
        __builtin_amdgcn_sched_group_barrier(0x008, 1, 0);
        __builtin_amdgcn_sched_group_barrier(0x100, 2, 0);
        __builtin_amdgcn_sched_group_barrier(0x008, 1, 0);
        __builtin_amdgcn_sched_group_barrier(0x100, 2, 0);
        __builtin_amdgcn_sched_group_barrier(0x008, 1, 0);
        __builtin_amdgcn_sched_group_barrier(0x100, 2, 0);
        __builtin_amdgcn_sched_group_barrier(0x008, 1, 0);
        __builtin_amdgcn_sched_group_barrier(0x100, 2, 0);
        __builtin_amdgcn_sched_group_barrier(0x008, 1, 0);
        __builtin_amdgcn_sched_group_barrier(0x100, 2, 0);
    }

    // ---- Accumulate partials.
    // D layout: VGPR r -> row r (+8 for hi lanes), col = lane&15.
    #pragma unroll
    for (int mt = 0; mt < MTILES; ++mt) {
        #pragma unroll
        for (int r = 0; r < 8; ++r) {
            const int m = mt * 16 + r + hiK * 8;
            if (m < BATCH)
                unsafeAtomicAdd(&Cp[(size_t)m * N + cn], acc[mt][r]);
        }
    }
}

// h_bf16[MPAD][N] = relu(partial + bias), zero rows >= BATCH.
__global__ void bias_relu_bf16_kernel(const float* __restrict__ part,
                                      const float* __restrict__ bias,
                                      __bf16* __restrict__ h, int nshift)
{
    const int idx = blockIdx.x * blockDim.x + threadIdx.x;   // [0, MPAD << nshift)
    const int N = 1 << nshift;
    const int m = idx >> nshift;
    const int j = idx & (N - 1);
    float v = 0.0f;
    if (m < BATCH) {
        v = part[m * N + j] + bias[j];
        v = v > 0.0f ? v : 0.0f;
    }
    h[idx] = (__bf16)v;
}

// zb_bf16[MPAD][1024] = x (zero-padded rows), for the step-0 layer-0 GEMM.
__global__ void pad_bf16_kernel(const float* __restrict__ x, __bf16* __restrict__ zb)
{
    const int idx = blockIdx.x * blockDim.x + threadIdx.x;   // [0, MPAD*1024)
    const int m = idx >> 10;
    const int j = idx & (DIMZ - 1);
    zb[idx] = (__bf16)((m < BATCH) ? x[m * DIMZ + j] : 0.0f);
}

// z_next = z_cur - (partial + b3), optional flip; also maintains the padded
// bf16 mirror consumed by the next step's layer-0 GEMM.
__global__ void update_z_kernel(const float* __restrict__ zc,
                                const float* __restrict__ part,
                                const float* __restrict__ bias,
                                float* __restrict__ zn,
                                __bf16* __restrict__ znb, int flip)
{
    const int idx = blockIdx.x * blockDim.x + threadIdx.x;   // [0, MPAD*1024)
    const int m = idx >> 10;
    const int j = idx & (DIMZ - 1);
    if (m < BATCH) {
        const float v = zc[m * DIMZ + j] - (part[m * DIMZ + j] + bias[j]);
        const int jj = flip ? (DIMZ - 1 - j) : j;
        zn[m * DIMZ + jj]  = v;
        znb[m * DIMZ + jj] = (__bf16)v;
    } else {
        znb[idx] = (__bf16)0.0f;    // whole row is zero; index order irrelevant
    }
}

// out[i] = 0.5 * sum_j z[i,j]^2 + 0.5 * DIM * log(2*pi)
__global__ void logprob_kernel(const float* __restrict__ z, float* __restrict__ out)
{
    __shared__ float red[256];
    const int i = blockIdx.x;
    float s = 0.0f;
    for (int j = threadIdx.x; j < DIMZ; j += 256) {
        const float v = z[i * DIMZ + j];
        s += v * v;
    }
    red[threadIdx.x] = s;
    __syncthreads();
    for (int off = 128; off > 0; off >>= 1) {
        if (threadIdx.x < off) red[threadIdx.x] += red[threadIdx.x + off];
        __syncthreads();
    }
    if (threadIdx.x == 0)
        out[i] = 0.5f * red[0] + 0.5f * (float)DIMZ * 1.8378770664093453f;
}

// ---------------------------------------------------------------------------

template <int K, int N, int MASK>
static inline void launch_gemm(const void* Ab, const float* W, float* Cp,
                               hipStream_t stream)
{
    constexpr int Kc = K / KSPLIT;
    const dim3 grid(N / 128, KSPLIT);
    const size_t lds = (size_t)MPAD * (Kc + LDS_PAD) * 2;   // bf16 bytes
    hipMemsetAsync(Cp, 0, (size_t)BATCH * N * sizeof(float), stream);
    made_gemm_kernel<K, N, MASK><<<grid, dim3(256), lds, stream>>>(
        (const __bf16*)Ab, W, Cp);
}

extern "C" void kernel_launch(void* const* d_in, const int* in_sizes, int n_in,
                              void* d_out, int out_size, void* d_ws, size_t ws_size,
                              hipStream_t stream)
{
    (void)in_sizes; (void)n_in; (void)out_size; (void)ws_size;

    const float* x  = (const float*)d_in[0];
    const float* W0 = (const float*)d_in[1];
    const float* b0 = (const float*)d_in[2];
    const float* W1 = (const float*)d_in[3];
    const float* b1 = (const float*)d_in[4];
    const float* W2 = (const float*)d_in[5];
    const float* b2 = (const float*)d_in[6];
    const float* W3 = (const float*)d_in[7];
    const float* b3 = (const float*)d_in[8];

    // Workspace layout (byte offsets; bf16 buffers 16B-aligned).
    char* ws = (char*)d_ws;
    float* part = (float*)(ws);                                 // [100,2048] f32
    float* z0   = (float*)(ws + 819200);                        // [100,1024] f32
    float* z1   = (float*)(ws + 819200 + 409600);               // [100,1024] f32
    char*  hbA  =          ws + 819200 + 819200;                // [112,2048] bf16
    char*  hbB  =          hbA + (size_t)MPAD * HID * 2;        // [112,2048] bf16
    char*  zb   =          hbB + (size_t)MPAD * HID * 2;        // [112,1024] bf16

    const int ethreads = 256;
    const int gH = (MPAD * HID)  / ethreads;   // 896
    const int gD = (MPAD * DIMZ) / ethreads;   // 448

    const float* zcur = x;
    float* zf32[2] = { z0, z1 };

    for (int s = 0; s < 4; ++s) {
        const float* W0s = W0 + (size_t)s * DIMZ * HID;
        const float* W1s = W1 + (size_t)s * HID  * HID;
        const float* W2s = W2 + (size_t)s * HID  * HID;
        const float* W3s = W3 + (size_t)s * HID  * DIMZ;
        const float* b0s = b0 + (size_t)s * HID;
        const float* b1s = b1 + (size_t)s * HID;
        const float* b2s = b2 + (size_t)s * HID;
        const float* b3s = b3 + (size_t)s * DIMZ;

        if (s == 0)
            pad_bf16_kernel<<<gD, ethreads, 0, stream>>>(x, (__bf16*)zb);

        // layer 0: z(1024) -> h(2048), mask M0 (exclusive)
        launch_gemm<DIMZ, HID, 0>(zb, W0s, part, stream);
        bias_relu_bf16_kernel<<<gH, ethreads, 0, stream>>>(part, b0s, (__bf16*)hbA, 11);

        // layer 1: h(2048) -> h(2048), mask M1
        launch_gemm<HID, HID, 1>(hbA, W1s, part, stream);
        bias_relu_bf16_kernel<<<gH, ethreads, 0, stream>>>(part, b1s, (__bf16*)hbB, 11);

        // layer 2: h(2048) -> h(2048), mask M1
        launch_gemm<HID, HID, 1>(hbB, W2s, part, stream);
        bias_relu_bf16_kernel<<<gH, ethreads, 0, stream>>>(part, b2s, (__bf16*)hbA, 11);

        // layer 3: h(2048) -> shift(1024), mask M3; then z update (+flip)
        launch_gemm<HID, DIMZ, 3>(hbA, W3s, part, stream);
        float* znext = zf32[s & 1];
        update_z_kernel<<<gD, ethreads, 0, stream>>>(zcur, part, b3s, znext,
                                                     (__bf16*)zb, (s < 3) ? 1 : 0);
        zcur = znext;
    }

    logprob_kernel<<<BATCH, 256, 0, stream>>>(zcur, (float*)d_out);
}
